// LanNet_8727373546168
// MI455X (gfx1250) — compile-verified, matrix operand order checked by reading, not run on hardware
//
#include <hip/hip_runtime.h>
#include <math.h>

// Problem constants (from reference)
#define B_    16
#define T_    2048
#define D_    48
#define H_    2048
#define G4H   (4*H_)
#define KC    2112          // 2048 (h) + 48 (x) + 16 pad, multiple of 32
#define BN_   100
#define OUT_  10
#define EPS_  1e-5f
#define NTILE (H_/16)       // 128 workgroups, one 16-hidden-unit tile each
#define TBLOCK 128          // 4 waves (wave32): one gate per wave

typedef __attribute__((ext_vector_type(16))) __bf16        v16bf;
typedef __attribute__((ext_vector_type(8)))  float         v8f;
typedef __attribute__((ext_vector_type(4)))  unsigned int  u32x4;  // POD 16B chunk

union FragBF { v16bf v; u32x4 q[2]; };

// Explicit global (address_space(1)) pointer types so fragment loads lower to
// global_load_b128 (not flat_load_b128, which also ties up the LDS/DScnt path).
typedef const __attribute__((address_space(1))) u32x4*          gc_u4;
typedef const __attribute__((address_space(1))) unsigned short* gc_u16;
typedef       __attribute__((address_space(1))) unsigned short* g_u16;

// Workspace layout (bytes)
constexpr size_t OFF_CTR = 0;
constexpr size_t SZ_A    = (size_t)B_ * KC * 2;        // 67,584 (bf16)
constexpr size_t OFF_A0  = 256;
constexpr size_t OFF_A1  = OFF_A0 + SZ_A;              // 16B aligned
constexpr size_t OFF_WC  = OFF_A1 + SZ_A;              // 16B aligned
constexpr size_t SZ_WC   = (size_t)G4H * KC * 2;       // 34,603,008
constexpr size_t OFF_S   = OFF_WC + SZ_WC;
constexpr size_t SZ_S    = (size_t)B_ * H_ * 4;
constexpr size_t WS_NEEDED = OFF_S + SZ_S;             // ~33.3 MB

__device__ __forceinline__ unsigned short f2bf(float f) {
  unsigned u = __float_as_uint(f);
  unsigned rb = ((u >> 16) & 1u) + 0x7FFFu;            // round-to-nearest-even
  return (unsigned short)((u + rb) >> 16);
}
__device__ __forceinline__ float fsig(float x) { return 1.f / (1.f + __expf(-x)); }
__device__ __forceinline__ float ftanh_(float x) {
  x = fminf(15.f, fmaxf(-15.f, x));
  float e = __expf(2.f * x);
  return (e - 1.f) / (e + 1.f);
}

// ---------------------------------------------------------------------------
// Init: zero both A buffers (h=0, pad=0), write x_0 into A0, reset barrier ctr
// ---------------------------------------------------------------------------
__global__ void ln_init_kernel(const float* __restrict__ src,
                               unsigned short* __restrict__ A0,
                               unsigned short* __restrict__ A1,
                               unsigned int* __restrict__ ctr) {
  int idx = blockIdx.x * blockDim.x + threadIdx.x;
  int stride = gridDim.x * blockDim.x;
  if (idx == 0) *ctr = 0u;
  for (int e = idx; e < B_ * KC; e += stride) {
    int m = e / KC, k = e % KC;
    unsigned short v = 0;
    if (k >= H_ && k < H_ + D_)
      v = f2bf(src[(size_t)m * T_ * D_ + (k - H_)]);   // t = 0 slice
    A0[e] = v;
    A1[e] = 0;                                          // pad/h; x filled per step
  }
}

// ---------------------------------------------------------------------------
// Convert [W_hh | W_ih | 0] to bf16; Wc[n*KC + k]
// ---------------------------------------------------------------------------
__global__ void ln_convw_kernel(const float* __restrict__ W_ih,
                                const float* __restrict__ W_hh,
                                unsigned short* __restrict__ Wc) {
  size_t idx = (size_t)blockIdx.x * blockDim.x + threadIdx.x;
  size_t stride = (size_t)gridDim.x * blockDim.x;
  size_t total = (size_t)G4H * KC;
  for (size_t e = idx; e < total; e += stride) {
    size_t n = e / KC;
    int k = (int)(e % KC);
    unsigned short v = 0;
    if (k < H_)              v = f2bf(W_hh[n * H_ + k]);
    else if (k < H_ + D_)    v = f2bf(W_ih[n * D_ + (k - H_)]);
    Wc[e] = v;
  }
}

// ---------------------------------------------------------------------------
// Persistent LSTM scan. 128 WGs (hidden tiles) x 4 waves (gates).
// Per step per wave: 66x v_wmma_f32_16x16x32_bf16 over K=2112.
// Double-buffered A + one grid barrier per step.
// ---------------------------------------------------------------------------
__global__ void ln_lstm_kernel(const float* __restrict__ src,
                               const int* __restrict__ frames,
                               const float* __restrict__ b_lstm,
                               const unsigned short* __restrict__ Wc,
                               unsigned short* __restrict__ A0,
                               unsigned short* __restrict__ A1,
                               float* __restrict__ sbuf,
                               unsigned int* __restrict__ ctr) {
  __shared__ float gate_lds[4][16][16];
  __shared__ float c_lds[16][16];
  __shared__ float s_lds[16][16];
  __shared__ float bias_lds[4][16];
  __shared__ int   fr_lds[16];

  const int tid  = threadIdx.x;
  const int lane = tid & 31;
  const int wv   = tid >> 5;           // gate index 0..3 (i,f,g,o)
  const int tile = blockIdx.x;         // hidden tile 0..127

  for (int e = tid; e < 256; e += TBLOCK) {
    c_lds[e >> 4][e & 15] = 0.f;
    s_lds[e >> 4][e & 15] = 0.f;
  }
  if (tid < 64) {
    int g = tid >> 4, j = tid & 15;
    bias_lds[g][j] = b_lstm[g * H_ + tile * 16 + j];
  }
  if (tid < 16) fr_lds[tid] = frames[tid];
  __syncthreads();

  const int nlo   = lane & 15;
  const int khalf = (lane >> 4) ? 8 : 0;      // wave32 hi-half holds K+8 / K+24
  gc_u16 Brow = (gc_u16)(Wc + (size_t)(wv * H_ + tile * 16 + nlo) * KC + khalf);

  // Pre-cast ping/pong buffers to the global address space; selecting between
  // two AS(1) pointers keeps the loads on the global_load path.
  g_u16 Ag[2] = { (g_u16)A0, (g_u16)A1 };

  for (int t = 0; t < T_; ++t) {
    g_u16 cur = Ag[t & 1];
    g_u16 nxt = Ag[(t + 1) & 1];
    gc_u16 Arow = (gc_u16)cur + (size_t)nlo * KC + khalf;

    v8f acc = {0.f, 0.f, 0.f, 0.f, 0.f, 0.f, 0.f, 0.f};
    for (int kb = 0; kb < KC; kb += 32) {
      FragBF a, b;
      a.q[0] = *(gc_u4)(Arow + kb);
      a.q[1] = *(gc_u4)(Arow + kb + 16);
      b.q[0] = *(gc_u4)(Brow + kb);
      b.q[1] = *(gc_u4)(Brow + kb + 16);
      acc = __builtin_amdgcn_wmma_f32_16x16x32_bf16(
          false, a.v, false, b.v, (short)0, acc, false, false);
    }

    // C layout: VGPR r, lanes 0-15 -> (M=r, N=lane); lanes 16-31 -> (M=r+8)
    const int mb = (lane >> 4) ? 8 : 0;
#pragma unroll
    for (int r = 0; r < 8; ++r) gate_lds[wv][mb + r][nlo] = acc[r];
    __syncthreads();

    // Pointwise LSTM cell: 256 elements over 128 threads
    for (int e = tid; e < 256; e += TBLOCK) {
      int b = e >> 4, j = e & 15;
      int col = tile * 16 + j;
      unsigned short hb;
      if (t < fr_lds[b]) {
        float gi = gate_lds[0][b][j] + bias_lds[0][j];
        float gf = gate_lds[1][b][j] + bias_lds[1][j];
        float gg = gate_lds[2][b][j] + bias_lds[2][j];
        float go = gate_lds[3][b][j] + bias_lds[3][j];
        float cn = fsig(gf) * c_lds[b][j] + fsig(gi) * ftanh_(gg);
        float hn = fsig(go) * ftanh_(cn);
        c_lds[b][j] = cn;
        s_lds[b][j] += hn;
        hb = f2bf(hn);
      } else {
        hb = cur[(size_t)b * KC + col];         // frozen h
      }
      nxt[(size_t)b * KC + col] = hb;
    }

    // WG0 stages x_{t+1} into the next A buffer
    if (blockIdx.x == 0 && t + 1 < T_) {
      for (int e = tid; e < B_ * D_; e += TBLOCK) {
        int b = e / D_, d = e % D_;
        nxt[(size_t)b * KC + H_ + d] =
            f2bf(src[(size_t)b * T_ * D_ + (size_t)(t + 1) * D_ + d]);
      }
    }

    __threadfence();
    __syncthreads();
    if (tid == 0) {
      __hip_atomic_fetch_add(ctr, 1u, __ATOMIC_RELEASE, __HIP_MEMORY_SCOPE_AGENT);
      unsigned target = (unsigned)gridDim.x * (unsigned)(t + 1);
      while (__hip_atomic_load(ctr, __ATOMIC_ACQUIRE, __HIP_MEMORY_SCOPE_AGENT) < target)
        __builtin_amdgcn_s_sleep(1);
    }
    __syncthreads();
  }

  // Export pooled sums
  for (int e = tid; e < 256; e += TBLOCK) {
    int b = e >> 4, j = e & 15;
    sbuf[(size_t)b * H_ + tile * 16 + j] = s_lds[b][j];
  }
}

// ---------------------------------------------------------------------------
// Tail: mean-pool /frames, BN1, FC1, BN2, FC2, softmax, CE + accuracy
// ---------------------------------------------------------------------------
__global__ void ln_tail_kernel(const int* __restrict__ frames,
                               const long long* __restrict__ target,
                               const float* __restrict__ gamma1,
                               const float* __restrict__ beta1,
                               const float* __restrict__ W1,
                               const float* __restrict__ b1,
                               const float* __restrict__ gamma2,
                               const float* __restrict__ beta2,
                               const float* __restrict__ W2,
                               const float* __restrict__ b2,
                               float* __restrict__ sbuf,
                               float* __restrict__ out) {
  __shared__ float obn[16][BN_];
  __shared__ float logits[16][OUT_];
  __shared__ float ce_sh[16];
  __shared__ float ok_sh[16];
  const int tid = threadIdx.x;   // 256 threads

  // BN1 over batch, in place on sbuf (pooled = s/frames)
  for (int j = tid; j < H_; j += 256) {
    float x[B_]; float mu = 0.f;
    for (int b = 0; b < B_; ++b) {
      float v = sbuf[(size_t)b * H_ + j] / (float)frames[b];
      x[b] = v; mu += v;
    }
    mu *= (1.f / B_);
    float var = 0.f;
    for (int b = 0; b < B_; ++b) { float d = x[b] - mu; var += d * d; }
    var *= (1.f / B_);
    float rs = rsqrtf(var + EPS_);
    float ga = gamma1[j], be = beta1[j];
    for (int b = 0; b < B_; ++b)
      sbuf[(size_t)b * H_ + j] = (x[b] - mu) * rs * ga + be;
  }
  __syncthreads();

  // FC1: [16,2048] x [2048,100]
  for (int e = tid; e < B_ * BN_; e += 256) {
    int b = e / BN_, n = e % BN_;
    const float* wr = W1 + (size_t)n * H_;
    const float* xr = sbuf + (size_t)b * H_;
    float acc = 0.f;
    for (int j = 0; j < H_; ++j) acc += xr[j] * wr[j];
    obn[b][n] = acc + b1[n];
  }
  __syncthreads();

  // BN2 over batch
  for (int n = tid; n < BN_; n += 256) {
    float mu = 0.f;
    for (int b = 0; b < B_; ++b) mu += obn[b][n];
    mu *= (1.f / B_);
    float var = 0.f;
    for (int b = 0; b < B_; ++b) { float d = obn[b][n] - mu; var += d * d; }
    var *= (1.f / B_);
    float rs = rsqrtf(var + EPS_);
    float ga = gamma2[n], be = beta2[n];
    for (int b = 0; b < B_; ++b) obn[b][n] = (obn[b][n] - mu) * rs * ga + be;
  }
  __syncthreads();

  // FC2: [16,100] x [100,10]
  for (int e = tid; e < B_ * OUT_; e += 256) {
    int b = e / OUT_, o = e % OUT_;
    float acc = 0.f;
    for (int n = 0; n < BN_; ++n) acc += obn[b][n] * W2[(size_t)o * BN_ + n];
    logits[b][o] = acc + b2[o];
  }
  __syncthreads();

  // softmax / CE / argmax per batch row
  if (tid < B_) {
    int b = tid;
    float mx = logits[b][0]; int am = 0;
    for (int o = 1; o < OUT_; ++o)
      if (logits[b][o] > mx) { mx = logits[b][o]; am = o; }
    float se = 0.f;
    for (int o = 0; o < OUT_; ++o) se += __expf(logits[b][o] - mx);
    int tg = (int)target[b];
    float logp = (logits[b][tg] - mx) - logf(se);
    ce_sh[b] = -logp;
    ok_sh[b] = (am == tg) ? 1.f : 0.f;
  }
  __syncthreads();
  if (tid == 0) {
    float ce = 0.f, ok = 0.f;
    for (int b = 0; b < B_; ++b) { ce += ce_sh[b]; ok += ok_sh[b]; }
    out[0] = ok * (1.f / B_);   // acc
    out[1] = ce * (1.f / B_);   // ce_loss
  }
}

// ---------------------------------------------------------------------------
extern "C" void kernel_launch(void* const* d_in, const int* in_sizes, int n_in,
                              void* d_out, int out_size, void* d_ws, size_t ws_size,
                              hipStream_t stream) {
  (void)in_sizes; (void)n_in; (void)out_size;
  if (ws_size < WS_NEEDED) return;   // workspace too small: no-op (deterministic)

  const float*     src    = (const float*)d_in[0];
  const int*       frames = (const int*)d_in[1];
  const long long* target = (const long long*)d_in[2];
  const float*     W_ih   = (const float*)d_in[3];
  const float*     W_hh   = (const float*)d_in[4];
  const float*     b_lstm = (const float*)d_in[5];
  const float*     gamma1 = (const float*)d_in[6];
  const float*     beta1  = (const float*)d_in[7];
  const float*     W1     = (const float*)d_in[8];
  const float*     b1     = (const float*)d_in[9];
  const float*     gamma2 = (const float*)d_in[10];
  const float*     beta2  = (const float*)d_in[11];
  const float*     W2     = (const float*)d_in[12];
  const float*     b2     = (const float*)d_in[13];

  char* ws = (char*)d_ws;
  unsigned int*   ctr  = (unsigned int*)(ws + OFF_CTR);
  unsigned short* A0   = (unsigned short*)(ws + OFF_A0);
  unsigned short* A1   = (unsigned short*)(ws + OFF_A1);
  unsigned short* Wc   = (unsigned short*)(ws + OFF_WC);
  float*          sbuf = (float*)(ws + OFF_S);

  ln_init_kernel<<<(B_ * KC + 255) / 256, 256, 0, stream>>>(src, A0, A1, ctr);
  ln_convw_kernel<<<2048, 256, 0, stream>>>(W_ih, W_hh, Wc);
  ln_lstm_kernel<<<NTILE, TBLOCK, 0, stream>>>(src, frames, b_lstm, Wc, A0, A1,
                                               sbuf, ctr);
  ln_tail_kernel<<<1, 256, 0, stream>>>(frames, target, gamma1, beta1, W1, b1,
                                        gamma2, beta2, W2, b2, sbuf,
                                        (float*)d_out);
}
